// ConditionalCNF_66073776882123
// MI455X (gfx1250) — matrix operand order repaired and checked by reading, not compiled, not run on hardware
//
#include <hip/hip_runtime.h>
#include <hip/hip_bf16.h>

typedef __attribute__((ext_vector_type(16))) _Float16 v16h;
typedef __attribute__((ext_vector_type(8)))  float    v8f;

// gfx1250 V_TANH_F32 (confirmed lowering last round).
#if __has_builtin(__builtin_amdgcn_tanhf)
#define TANHF(x) __builtin_amdgcn_tanhf(x)
#elif __has_builtin(__builtin_amdgcn_tanh_f32)
#define TANHF(x) __builtin_amdgcn_tanh_f32(x)
#else
static __device__ __forceinline__ float TANHF(float x) {
  float e = __expf(2.0f * x);
  return 1.0f - 2.0f / (e + 1.0f);
}
#endif

// lane <-> lane^16 exchange. Prefer v_permlanex16_b32 (pure VALU, no DS wait);
// identity nibble selectors make lane i fetch lane i^16.
#if __has_builtin(__builtin_amdgcn_permlanex16)
static __device__ __forceinline__ float xchg_half(float v) {
  int i = __builtin_bit_cast(int, v);
  int r = __builtin_amdgcn_permlanex16(i, i, 0x76543210, 0xFEDCBA98, false, false);
  return __builtin_bit_cast(float, r);
}
#else
static __device__ __forceinline__ float xchg_half(float v) {
  return __shfl_xor(v, 16, 32);
}
#endif

// probe-confirmed 8-arg signature: (neg_a, A, neg_b, B, c_mod, C, reuse_a, reuse_b)
#define WMMA_F16(A, Bm, C) \
  __builtin_amdgcn_wmma_f32_16x16x32_f16(false, (A), false, (Bm), (short)0, (C), false, false)

// One wave handles 16 samples. D = A(16x32 f16) x B(32x16 f16) + C(16x16 f32)
//   A = W2 rows (constant, register-resident), B = h1^T / d1^T, D = h2^T / u^T.
// B layout: lane n (n<16) holds sample n features K=0..15; lane n+16 holds K=16..31.
// A layout: lane m row m (+16 for tile1); lane<16 cols 0..15, lane>=16 cols 16..31.
// D layout: v8f elem r: lanes0-15 -> feature r(+tile*16), lanes16-31 -> feature r+8(+tile*16).
__global__ __launch_bounds__(256) void cnf_rk4_wmma_kernel(
    const float* __restrict__ T,    const float* __restrict__ cond,
    const float* __restrict__ W1,   const float* __restrict__ b1,
    const float* __restrict__ W2,   const float* __restrict__ b2,
    const float* __restrict__ W3,   const float* __restrict__ b3,
    float* __restrict__ out, int B)
{
  __shared__ float sW1[32 * 9];
  __shared__ float sW2[32 * 32];
  __shared__ float sb1[32];
  __shared__ float sb2[32];
  __shared__ float sW3[32];
  __shared__ float sb3v[1];

  for (int i = threadIdx.x; i < 32 * 9;  i += blockDim.x) sW1[i] = W1[i];
  for (int i = threadIdx.x; i < 32 * 32; i += blockDim.x) sW2[i] = W2[i];
  if (threadIdx.x < 32) {
    sb1[threadIdx.x] = b1[threadIdx.x];
    sb2[threadIdx.x] = b2[threadIdx.x];
    sW3[threadIdx.x] = W3[threadIdx.x];
  }
  if (threadIdx.x == 0) sb3v[0] = b3[0];
  __syncthreads();

  const int lane   = threadIdx.x & 31;
  const int wave   = threadIdx.x >> 5;
  const int laneHi = lane >> 4;   // 0: lanes 0-15, 1: lanes 16-31
  const int laneLo = lane & 15;

  const int sample = (blockIdx.x * (blockDim.x >> 5) + wave) * 16 + laneLo;
  const int sl     = sample < B ? sample : (B - 1);   // branchless clamp (keeps EXEC full)

  float cnd[8];
#pragma unroll
  for (int c = 0; c < 8; ++c) cnd[c] = cond[sl * 8 + c];

  float z  = T[sl];
  float lp = 0.0f;

  // ---- hoisted cond contribution + per-lane constant vectors -------------
  // B-layout features: f = laneHi*16 + i
  float w1k0b[16], c1b[16];
#pragma unroll
  for (int i = 0; i < 16; ++i) {
    const int f = laneHi * 16 + i;
    w1k0b[i] = sW1[f * 9 + 0];
    float a = sb1[f];
#pragma unroll
    for (int c = 0; c < 8; ++c) a = fmaf(sW1[f * 9 + 1 + c], cnd[c], a);
    c1b[i] = a;
  }

  // D-layout features: tile0 -> laneHi*8 + r, tile1 -> 16 + laneHi*8 + r
  float w3d0[8], w3d1[8];
  v8f cb2_0, cb2_1;
#pragma unroll
  for (int r = 0; r < 8; ++r) {
    const int f0 = laneHi * 8 + r;
    const int f1 = 16 + f0;
    w3d0[r]  = sW3[f0];
    w3d1[r]  = sW3[f1];
    cb2_0[r] = sb2[f0];   // bias pre-baked into WMMA C operand
    cb2_1[r] = sb2[f1];
  }
  const float b3s = sb3v[0];

  // A tiles: W2 in f16, resident for the whole integration
  v16h aW2_0, aW2_1;
  {
    const int row = laneLo;
    const int kb  = laneHi * 16;
#pragma unroll
    for (int j = 0; j < 8; ++j) {
      aW2_0[j]     = (_Float16)sW2[row * 32 + kb + j];
      aW2_0[8 + j] = (_Float16)sW2[row * 32 + 8 + kb + j];
      aW2_1[j]     = (_Float16)sW2[(row + 16) * 32 + kb + j];
      aW2_1[8 + j] = (_Float16)sW2[(row + 16) * 32 + 8 + kb + j];
    }
  }

  const v8f czero = {};
  const float dt = (1.0f - 0.0f) / 20.0f;

  // ---- one dynamics evaluation: (dz/dt, divergence) ----------------------
  auto evalDyn = [&](float zc, float& dzdt, float& divg) {
    // h1 = tanh(z*W1[:,0] + c1) and d1 = (1-h1^2)*W1[:,0], both in B layout
    float h1[16];
    v16h bh1, bd1;
#pragma unroll
    for (int i = 0; i < 16; ++i) {
      const float t = TANHF(fmaf(zc, w1k0b[i], c1b[i]));
      h1[i]  = t;
      bh1[i] = (_Float16)t;
    }
#pragma unroll
    for (int i = 0; i < 16; ++i) {
      const float t = fmaf(-h1[i], h1[i], 1.0f) * w1k0b[i];
      bd1[i] = (_Float16)t;
    }

    // All four WMMAs are mutually independent: issue back-to-back.
    v8f p0 = WMMA_F16(aW2_0, bh1, cb2_0);   // pre2 = W2*h1 + b2
    v8f p1 = WMMA_F16(aW2_1, bh1, cb2_1);
    v8f u0 = WMMA_F16(aW2_0, bd1, czero);   // u = W2*d1
    v8f u1 = WMMA_F16(aW2_1, bd1, czero);

    float h2a[8], h2b[8];
#pragma unroll
    for (int r = 0; r < 8; ++r) {
      h2a[r] = TANHF(p0[r]);
      h2b[r] = TANHF(p1[r]);
    }

    // dual accumulators per sum -> shorter dep chains, dual-issue friendly
    float s0 = 0.0f, s1 = 0.0f, v0 = 0.0f, v1 = 0.0f;
#pragma unroll
    for (int r = 0; r < 8; ++r) {
      s0 = fmaf(w3d0[r], h2a[r], s0);
      s1 = fmaf(w3d1[r], h2b[r], s1);
      const float d2a = fmaf(-h2a[r], h2a[r], 1.0f) * u0[r];
      const float d2b = fmaf(-h2b[r], h2b[r], 1.0f) * u1[r];
      v0 = fmaf(w3d0[r], d2a, v0);
      v1 = fmaf(w3d1[r], d2b, v1);
    }
    const float s  = s0 + s1;
    const float sv = v0 + v1;

    // single exchange point for both reductions (VALU permlanex16, no DS wait)
    const float so  = xchg_half(s);
    const float svo = xchg_half(sv);
    dzdt = (s + so) + b3s;
    divg = sv + svo;
  };

  // ---- RK4, 20 steps ------------------------------------------------------
#pragma unroll 1
  for (int st = 0; st < 20; ++st) {
    float k1z, k1l, k2z, k2l, k3z, k3l, k4z, k4l;
    evalDyn(z, k1z, k1l);
    evalDyn(fmaf(0.5f * dt, k1z, z), k2z, k2l);
    evalDyn(fmaf(0.5f * dt, k2z, z), k3z, k3l);
    evalDyn(fmaf(dt,        k3z, z), k4z, k4l);
    z  = fmaf(dt / 6.0f, k1z + 2.0f * (k2z + k3z) + k4z, z);
    lp = fmaf(dt / 6.0f, k1l + 2.0f * (k2l + k3l) + k4l, lp);
  }

  // out = [z_T (B), delta_logp (B)]. Both lane halves hold identical (z, lp):
  // lanes 0-15 store z_T, lanes 16-31 store delta_logp -> one full-wave store.
  if (sample < B) {
    out[(laneHi ? B : 0) + sample] = laneHi ? lp : z;
  }
}

extern "C" void kernel_launch(void* const* d_in, const int* in_sizes, int n_in,
                              void* d_out, int out_size, void* d_ws, size_t ws_size,
                              hipStream_t stream) {
  const float* T    = (const float*)d_in[0];
  const float* cond = (const float*)d_in[1];
  const float* W1   = (const float*)d_in[2];
  const float* b1   = (const float*)d_in[3];
  const float* W2   = (const float*)d_in[4];
  const float* b2   = (const float*)d_in[5];
  const float* W3   = (const float*)d_in[6];
  const float* b3   = (const float*)d_in[7];
  float* out = (float*)d_out;

  const int B = in_sizes[0];                     // T is [B,1]
  const int samplesPerBlock = (256 / 32) * 16;   // 8 waves x 16 samples = 128
  const int grid = (B + samplesPerBlock - 1) / samplesPerBlock;

  cnf_rk4_wmma_kernel<<<grid, 256, 0, stream>>>(T, cond, W1, b1, W2, b2, W3, b3, out, B);
}
